// SharedOccHead_46136538693795
// MI455X (gfx1250) — matrix operand order repaired and checked by reading, not compiled, not run on hardware
//
#include <hip/hip_runtime.h>
#include <hip/hip_bf16.h>

typedef __attribute__((ext_vector_type(16))) __bf16 v16bf;
typedef __attribute__((ext_vector_type(8)))  __bf16 v8bf;
typedef __attribute__((ext_vector_type(8)))  float  v8f;

#define NPTS   40000
#define NCAM   6
#define CIMG   512
#define HF     56
#define WF     100
#define HW     (HF*WF)        // 5600
#define CCH    128
#define NVOX   262144         // 128*128*16
#define OUTC   17
#define NROWS  (NVOX+NPTS)
#define NTILES (NROWS/16)     // 18884

union AFrag { v16bf v; v8bf h[2]; __bf16 e[16]; };

static __device__ __forceinline__ v8f wmma_bf16(v16bf a, v16bf b, v8f c) {
  // (neg_a, A, neg_b, B, c_mod, C, reuse_a, reuse_b)
  return __builtin_amdgcn_wmma_f32_16x16x32_bf16(false, a, false, b, (short)0, c, false, false);
}

// A-operand fragment from row-major [16][K] (rowStride in halves), k-chunk kb.
// 16-bit A layout: lanes 0-15: K = 0..7 then 16..23 (for this 32-chunk); lanes 16-31: +8.
static __device__ __forceinline__ v16bf load_a_frag(const __bf16* base, int rowStride,
                                                    int m, int h, int kb) {
  AFrag f;
  const __bf16* r = base + m*rowStride + kb + h*8;
  f.h[0] = *(const v8bf*)(r);
  f.h[1] = *(const v8bf*)(r + 16);
  return f.v;
}

// B-operand fragment from row-major [K][N] (rowStride in halves), k-chunk kb, col n0.
// 16-bit B layout: lane l -> N = n0 + l%16 ; elements -> K = kb + 8h + 0..7 then kb + 16 + 8h + 0..7.
static __device__ __forceinline__ v16bf load_b_frag(const __bf16* base, int rowStride,
                                                    int lane, int kb, int n0) {
  AFrag f;
  const __bf16* r = base + (size_t)(kb + (lane >> 4)*8)*rowStride + n0 + (lane & 15);
  for (int s = 0; s < 8; ++s) {
    f.e[s]     = r[(size_t)s*rowStride];
    f.e[8 + s] = r[(size_t)(16 + s)*rowStride];
  }
  return f.v;
}

// ---------------- kernel 1: weights -> bf16 ----------------
__global__ void wcvt_kernel(const float* Wred, const float* W1, const float* W2, const float* W3,
                            __bf16* wr, __bf16* w1, __bf16* w2, __bf16* w3) {
  int i  = blockIdx.x*blockDim.x + threadIdx.x;
  int st = gridDim.x*blockDim.x;
  for (int k = i; k < CIMG*CCH; k += st) wr[k] = (__bf16)Wred[k];
  for (int k = i; k < CCH*CCH;  k += st) { w1[k] = (__bf16)W1[k]; w2[k] = (__bf16)W2[k]; }
  for (int k = i; k < CCH*32;   k += st) {
    int c = k >> 5, n = k & 31;
    w3[k] = (__bf16)((n < OUTC) ? W3[c*OUTC + n] : 0.f);
  }
}

// ---------------- kernel 2: red = img @ W_red + b_red ----------------
__global__ void __launch_bounds__(256) reduce_gemm_kernel(const float* __restrict__ img,
                                                          const __bf16* __restrict__ wr,
                                                          const float* __restrict__ b_red,
                                                          float* __restrict__ red) {
  __shared__ __attribute__((aligned(16))) __bf16 As[16*520];   // 16 pixels x 512 K (+pad)
  const int tid = threadIdx.x, lane = tid & 31, wave = tid >> 5;
  const int m = lane & 15, h = lane >> 4;
  for (int t = blockIdx.x; t < NCAM*(HW/16); t += gridDim.x) {
    const int cam = t / (HW/16);
    const int p0  = (t % (HW/16)) * 16;
    { // stage A tile (transpose [c][p] -> [p][c]) as bf16
      const int j = tid & 15;
      for (int c = tid >> 4; c < CIMG; c += 16)
        As[j*520 + c] = (__bf16)img[((size_t)cam*CIMG + c)*HW + p0 + j];
    }
    __syncthreads();
    const int n0 = wave * 16;
    const float bias = b_red[n0 + m];
    v8f acc; for (int r = 0; r < 8; ++r) acc[r] = bias;
    for (int kb = 0; kb < CIMG; kb += 32) {
      v16bf a = load_a_frag(As, 520, m, h, kb);
      v16bf b = load_b_frag(wr, CCH, lane, kb, n0);
      acc = wmma_bf16(a, b, acc);
    }
    for (int r = 0; r < 8; ++r) {
      const int p = p0 + h*8 + r;
      red[((size_t)cam*HW + p)*CCH + n0 + m] = acc[r];
    }
    __syncthreads();
  }
}

// ---------------- kernel 3: fused trilinear + masked bilinear sampling ----------------
__global__ void __launch_bounds__(256) sample_kernel(const float* __restrict__ vox,
                                                     const float* __restrict__ pts,
                                                     const float* __restrict__ puv,
                                                     const float* __restrict__ red,
                                                     float* __restrict__ point_feats,
                                                     float* __restrict__ img_add) {
  const int c = threadIdx.x & 127;
  const int n = blockIdx.x*2 + (threadIdx.x >> 7);
  if (n >= NPTS) return;

  // ---- trilinear on feat [C, D=X(128), H=Y(128), W=Z(16)]; x->W, y->H, z->D
  const float px = pts[n*3+0], py = pts[n*3+1], pz = pts[n*3+2];
  const float nx = (px + 51.2f)*(2.0f/102.4f) - 1.f;
  const float ny = (py + 51.2f)*(2.0f/102.4f) - 1.f;
  const float nz = (pz +  5.0f)*(2.0f/  8.0f) - 1.f;
  const float fx = (nx + 1.f)*8.f  - 0.5f;   // W = 16
  const float fy = (ny + 1.f)*64.f - 0.5f;   // H = 128
  const float fz = (nz + 1.f)*64.f - 0.5f;   // D = 128
  const float x0f = floorf(fx), y0f = floorf(fy), z0f = floorf(fz);
  const float tx = fx - x0f, ty = fy - y0f, tz = fz - z0f;
  const int x0 = (int)x0f, y0 = (int)y0f, z0 = (int)z0f;
  const float* vc = vox + (size_t)c * NVOX;
  float acc = 0.f;
  #pragma unroll
  for (int dz = 0; dz < 2; ++dz)
  #pragma unroll
  for (int dy = 0; dy < 2; ++dy)
  #pragma unroll
  for (int dx = 0; dx < 2; ++dx) {
    const int iw = x0+dx, ih = y0+dy, id = z0+dz;
    const bool valid = (iw>=0)&&(iw<16)&&(ih>=0)&&(ih<128)&&(id>=0)&&(id<128);
    const int cw = min(max(iw,0),15), chh = min(max(ih,0),127), cd = min(max(id,0),127);
    const float w = (dx?tx:1.f-tx)*(dy?ty:1.f-ty)*(dz?tz:1.f-tz);
    acc += vc[cd*2048 + chh*16 + cw] * w * (valid ? 1.f : 0.f);
  }
  point_feats[(size_t)n*CCH + c] = acc;

  // ---- masked bilinear over cams on red [cam][h][w][c]
  float ia = 0.f;
  for (int cam = 0; cam < NCAM; ++cam) {
    const float u  = puv[((size_t)n*NCAM + cam)*3 + 0];
    const float v_ = puv[((size_t)n*NCAM + cam)*3 + 1];
    const float dd = puv[((size_t)n*NCAM + cam)*3 + 2];
    const bool mask = (dd > 1e-5f) && (u > -1.f) && (u < 1.f) && (v_ > -1.f) && (v_ < 1.f);
    if (!mask) continue;
    const float gx = (u  + 1.f)*50.f - 0.5f;  // W = 100
    const float gy = (v_ + 1.f)*28.f - 0.5f;  // H = 56
    const float gx0 = floorf(gx), gy0 = floorf(gy);
    const float bx = gx - gx0, by = gy - gy0;
    const int ix0 = (int)gx0, iy0 = (int)gy0;
    const float* rc = red + (size_t)cam*HW*CCH + c;
    #pragma unroll
    for (int dy = 0; dy < 2; ++dy)
    #pragma unroll
    for (int dx = 0; dx < 2; ++dx) {
      const int ix = ix0+dx, iy = iy0+dy;
      const bool vld = (ix>=0)&&(ix<WF)&&(iy>=0)&&(iy<HF);
      const int cx = min(max(ix,0),WF-1), cy = min(max(iy,0),HF-1);
      const float w = (dx?bx:1.f-bx)*(dy?by:1.f-by);
      ia += rc[(size_t)(cy*WF + cx)*CCH] * w * (vld ? 1.f : 0.f);
    }
  }
  img_add[(size_t)n*CCH + c] = ia;
}

// ---------------- kernel 4: 3-layer MLP over 302144 rows ----------------
__global__ void __launch_bounds__(256) mlp_kernel(const float* __restrict__ vox,
                                                  const float* __restrict__ point_feats,
                                                  const float* __restrict__ img_add,
                                                  const __bf16* __restrict__ w1,
                                                  const __bf16* __restrict__ w2,
                                                  const __bf16* __restrict__ w3,
                                                  const float* __restrict__ b1,
                                                  const float* __restrict__ b2,
                                                  const float* __restrict__ b3,
                                                  float* __restrict__ dout) {
  __shared__ __attribute__((aligned(16))) __bf16 Hs[8][16*136];  // per-wave staging
  const int lane = threadIdx.x & 31, wave = threadIdx.x >> 5;
  const int m = lane & 15, h = lane >> 4;
  __bf16* hs = Hs[wave];

  for (int t = blockIdx.x*8 + wave; t < NTILES; t += gridDim.x*8) {
    const int row0 = t * 16;
    v16bf A[4];
    // ---- build joint A fragments (bf16) ----
    if (row0 < NVOX) {
      const int v = row0 + m;
      const bool addimg = (row0 < NPTS);                 // reference adds img_add to rows [0, N)
      const float* ia = img_add + (size_t)v * CCH;
      #pragma unroll
      for (int kf = 0; kf < 4; ++kf) {
        AFrag f;
        #pragma unroll
        for (int e = 0; e < 8; ++e) {
          const int c0 = kf*32 + h*8 + e;
          const int c1 = c0 + 16;
          float a0 = vox[(size_t)c0*NVOX + v];           // vox_flat[v][c] = voxel_feats[c][v]
          float a1 = vox[(size_t)c1*NVOX + v];
          if (addimg) { a0 += ia[c0]; a1 += ia[c1]; }
          f.e[e]     = (__bf16)a0;
          f.e[8 + e] = (__bf16)a1;
        }
        A[kf] = f.v;
      }
    } else {
      const float* src = point_feats + (size_t)(row0 - NVOX + m) * CCH;
      #pragma unroll
      for (int kf = 0; kf < 4; ++kf) {
        AFrag f;
        #pragma unroll
        for (int e = 0; e < 8; ++e) {
          f.e[e]     = (__bf16)src[kf*32 + h*8 + e];
          f.e[8 + e] = (__bf16)src[kf*32 + 16 + h*8 + e];
        }
        A[kf] = f.v;
      }
    }
    // ---- layer 1 (+softplus), C-layout -> LDS (A row-major) ----
    #pragma unroll
    for (int nt = 0; nt < 8; ++nt) {
      const int n0 = nt*16;
      const float bias = b1[n0 + m];
      v8f acc; for (int r = 0; r < 8; ++r) acc[r] = bias;
      #pragma unroll
      for (int kf = 0; kf < 4; ++kf)
        acc = wmma_bf16(A[kf], load_b_frag(w1, CCH, lane, kf*32, n0), acc);
      #pragma unroll
      for (int r = 0; r < 8; ++r) {
        const float x = acc[r];
        const float sp = fmaxf(x, 0.f) + log1pf(expf(-fabsf(x)));  // stable softplus
        hs[(h*8 + r)*136 + n0 + m] = (__bf16)sp;
      }
    }
    asm volatile("s_wait_dscnt 0" ::: "memory");
    #pragma unroll
    for (int kf = 0; kf < 4; ++kf) A[kf] = load_a_frag(hs, 136, m, h, kf*32);
    // ---- layer 2 (linear) ----
    #pragma unroll
    for (int nt = 0; nt < 8; ++nt) {
      const int n0 = nt*16;
      const float bias = b2[n0 + m];
      v8f acc; for (int r = 0; r < 8; ++r) acc[r] = bias;
      #pragma unroll
      for (int kf = 0; kf < 4; ++kf)
        acc = wmma_bf16(A[kf], load_b_frag(w2, CCH, lane, kf*32, n0), acc);
      #pragma unroll
      for (int r = 0; r < 8; ++r)
        hs[(h*8 + r)*136 + n0 + m] = (__bf16)acc[r];
    }
    asm volatile("s_wait_dscnt 0" ::: "memory");
    #pragma unroll
    for (int kf = 0; kf < 4; ++kf) A[kf] = load_a_frag(hs, 136, m, h, kf*32);
    // ---- layer 3 (128 -> 17, padded to 32) ----
    float outv[2][8];
    #pragma unroll
    for (int nt = 0; nt < 2; ++nt) {
      const int n0 = nt*16;
      const int col = n0 + m;
      const float bias = (col < OUTC) ? b3[col] : 0.f;
      v8f acc; for (int r = 0; r < 8; ++r) acc[r] = bias;
      #pragma unroll
      for (int kf = 0; kf < 4; ++kf)
        acc = wmma_bf16(A[kf], load_b_frag(w3, 32, lane, kf*32, n0), acc);
      #pragma unroll
      for (int r = 0; r < 8; ++r) outv[nt][r] = acc[r];
    }
    // ---- stage output tile [16][32] fp32 in LDS, then coalesced global stores ----
    float* os = (float*)hs;
    #pragma unroll
    for (int nt = 0; nt < 2; ++nt)
      #pragma unroll
      for (int r = 0; r < 8; ++r)
        os[(h*8 + r)*32 + nt*16 + m] = outv[nt][r];
    asm volatile("s_wait_dscnt 0" ::: "memory");
    if (row0 < NVOX) {
      // output_voxels[o][v] = out[v][o]
      for (int idx = lane; idx < OUTC*16; idx += 32) {
        const int o = idx >> 4, rr = idx & 15;
        dout[(size_t)o*NVOX + row0 + rr] = os[rr*32 + o];
      }
    } else {
      const size_t base = (size_t)OUTC*NVOX + (size_t)(row0 - NVOX)*OUTC;
      for (int idx = lane; idx < 16*OUTC; idx += 32)
        dout[base + idx] = os[(idx/OUTC)*32 + (idx%OUTC)];
    }
  }
}

extern "C" void kernel_launch(void* const* d_in, const int* in_sizes, int n_in,
                              void* d_out, int out_size, void* d_ws, size_t ws_size,
                              hipStream_t stream) {
  const float* vox  = (const float*)d_in[0];
  const float* pts  = (const float*)d_in[1];
  const float* img  = (const float*)d_in[2];
  const float* puv  = (const float*)d_in[3];
  const float* Wred = (const float*)d_in[4];
  const float* bred = (const float*)d_in[5];
  const float* W1   = (const float*)d_in[6];
  const float* b1   = (const float*)d_in[7];
  const float* W2   = (const float*)d_in[8];
  const float* b2   = (const float*)d_in[9];
  const float* W3   = (const float*)d_in[10];
  const float* b3   = (const float*)d_in[11];
  float* dout = (float*)d_out;

  char* ws = (char*)d_ws;
  size_t off = 0;
  float* red         = (float*)(ws + off); off += (size_t)NCAM*HW*CCH*sizeof(float);   // 17.2 MB
  float* point_feats = (float*)(ws + off); off += (size_t)NPTS*CCH*sizeof(float);      // 20.5 MB
  float* img_add     = (float*)(ws + off); off += (size_t)NPTS*CCH*sizeof(float);      // 20.5 MB
  __bf16* wrb = (__bf16*)(ws + off); off += (size_t)CIMG*CCH*2;
  __bf16* w1b = (__bf16*)(ws + off); off += (size_t)CCH*CCH*2;
  __bf16* w2b = (__bf16*)(ws + off); off += (size_t)CCH*CCH*2;
  __bf16* w3b = (__bf16*)(ws + off); off += (size_t)CCH*32*2;

  wcvt_kernel<<<100, 256, 0, stream>>>(Wred, W1, W2, W3, wrb, w1b, w2b, w3b);
  reduce_gemm_kernel<<<300, 256, 0, stream>>>(img, wrb, bred, red);
  sample_kernel<<<NPTS/2, 256, 0, stream>>>(vox, pts, puv, red, point_feats, img_add);
  mlp_kernel<<<512, 256, 0, stream>>>(vox, point_feats, img_add, w1b, w2b, w3b,
                                      b1, b2, b3, dout);
}
